// KnowledgeAttention_35905926595170
// MI455X (gfx1250) — matrix operand order repaired and checked by reading, not compiled, not run on hardware
//
#include <hip/hip_runtime.h>
#include <hip/hip_bf16.h>
#include <cmath>

// Problem constants (from reference)
#define BB 4
#define TT 512
#define KK 256
#define DD 256
#define UU 128

typedef __attribute__((ext_vector_type(16))) __bf16 v16bf;
typedef __attribute__((ext_vector_type(8)))  float  v8f;

__device__ __forceinline__ float fast_exp2(float x) {
#if __has_builtin(__builtin_amdgcn_exp2f)
  return __builtin_amdgcn_exp2f(x);
#else
  return exp2f(x);
#endif
}
__device__ __forceinline__ float fast_rcp(float x) {
#if __has_builtin(__builtin_amdgcn_rcpf)
  return __builtin_amdgcn_rcpf(x);
#else
  return 1.0f / x;
#endif
}
__device__ __forceinline__ float fast_tanh(float x) {
#if __has_builtin(__builtin_amdgcn_tanhf)
  return __builtin_amdgcn_tanhf(x);           // v_tanh_f32 (1 trans op)
#else
  float ax = __builtin_fabsf(x);
  float z  = fast_exp2(ax * -2.885390081777927f);   // exp(-2|x|)
  float r  = (1.0f - z) * fast_rcp(1.0f + z);
  return __builtin_copysignf(r, x);
#endif
}
// f32 -> bf16, round-to-nearest-even
__device__ __forceinline__ unsigned short f2bf(float f) {
  unsigned int u = __float_as_uint(f);
  u += 0x7FFFu + ((u >> 16) & 1u);
  return (unsigned short)(u >> 16);
}

// -------------------------------------------------------------------------
// Projection: Y[r,u] = X[r,:] . W[:,u] + bias[u]
// -------------------------------------------------------------------------
__global__ __launch_bounds__(256) void proj_kernel(
    const float* __restrict__ X, const float* __restrict__ W,
    const float* __restrict__ bias, float* __restrict__ Y,
    int rows, int din) {
  int idx = blockIdx.x * 256 + threadIdx.x;
  int r = idx / UU;
  int u = idx - r * UU;
  if (r >= rows) return;
  const float* x = X + (size_t)r * din;
  float acc = bias[u];
#pragma unroll 8
  for (int d = 0; d < din; ++d)
    acc = fmaf(x[d], W[d * UU + u], acc);
  Y[idx] = acc;
}

// -------------------------------------------------------------------------
// Repack knowledge f32 [B,K,D] -> bf16 B-fragment-major words:
//   word[((((b*8+kk)*16+nt)*2+g)*16+n16)*8+vv] = pack(kn[b][kb][d], kn[b][kb+1][d])
//   with kb = kk*32 + g*16 + 2*vv, d = nt*16 + n16.
// Each lane of the GEMM stage then loads its 8 words contiguously (2x b128).
// -------------------------------------------------------------------------
__global__ __launch_bounds__(256) void repack_kernel(
    const float* __restrict__ knowledge, unsigned int* __restrict__ kp) {
  int w = blockIdx.x * 256 + threadIdx.x;     // 0 .. BB*8*16*2*16*8-1
  int vv  = w & 7;
  int n16 = (w >> 3) & 15;
  int g   = (w >> 7) & 1;
  int nt  = (w >> 8) & 15;
  int kk  = (w >> 12) & 7;
  int b   = w >> 15;
  int kb = kk * 32 + g * 16 + vv * 2;
  int d  = nt * 16 + n16;
  float x0 = knowledge[((size_t)(b * KK + kb)    ) * DD + d];
  float x1 = knowledge[((size_t)(b * KK + kb + 1)) * DD + d];
  kp[w] = (unsigned int)f2bf(x0) | ((unsigned int)f2bf(x1) << 16);
}

// -------------------------------------------------------------------------
// Fused: scores (tanh-MLP) -> softmax -> context GEMM via bf16 WMMA.
// One block = one batch b and one 16-row tile of T. 256 threads = 8 waves.
// -------------------------------------------------------------------------
__global__ __launch_bounds__(256) void attn_fused_kernel(
    const unsigned int* __restrict__ kpack, // packed bf16 knowledge fragments
    const float* __restrict__ e_ws,         // [B,T,U] f32
    const float* __restrict__ k_ws,         // [B,K,U] f32
    const float* __restrict__ Vvec,         // [U]
    const float* __restrict__ bV,           // [1]
    float* __restrict__ out)                // [B,T,D] f32
{
  __shared__ float          e_lds[16][UU];   // 8 KB
  __shared__ float          k_lds[16][UU];   // 8 KB
  __shared__ float          sc[16][KK];      // 16 KB
  __shared__ unsigned short wbf[16][KK];     // 8 KB
  __shared__ float          v_lds[UU];
  __shared__ float          red[16][16];     // reduction scratch
  __shared__ float          rowstat[16];

  const int tid = threadIdx.x;
  const int b   = blockIdx.x / (TT / 16);
  const int t0  = (blockIdx.x % (TT / 16)) * 16;

  if (tid < UU) v_lds[tid] = Vvec[tid];
  {
    const float4* src = (const float4*)(e_ws + ((size_t)(b * TT + t0)) * UU);
    float4* dst = (float4*)&e_lds[0][0];
    for (int i = tid; i < 16 * UU / 4; i += 256) dst[i] = src[i];
  }
  __syncthreads();

  // ---- Stage 1: additive-attention scores with tanh ---------------------
  const int tl = tid & 15;   // t row within tile
  const int kl = tid >> 4;   // k row within 16-wide k tile
  const float bv = bV[0];
  for (int j = 0; j < KK / 16; ++j) {
    {
      const float4* src = (const float4*)(k_ws + ((size_t)(b * KK + j * 16)) * UU);
      float4* dst = (float4*)&k_lds[0][0];
      for (int i = tid; i < 16 * UU / 4; i += 256) dst[i] = src[i];
    }
    __syncthreads();
    float s = 0.f;
#pragma unroll 8
    for (int u = 0; u < UU; ++u) {
      float h = fast_tanh(e_lds[tl][u] + k_lds[kl][u]);
      s = fmaf(h, v_lds[u], s);
    }
    sc[tl][j * 16 + kl] = s + bv;
    __syncthreads();
  }

  // ---- Stage 2: softmax over K (two-level parallel reduction) -----------
  const float l2e = 1.4426950408889634f;
  {
    float pm = -3.0e38f;
    for (int k2 = kl * 16; k2 < kl * 16 + 16; ++k2) pm = fmaxf(pm, sc[tl][k2]);
    red[tl][kl] = pm;
  }
  __syncthreads();
  if (tid < 16) {
    float m = red[tid][0];
#pragma unroll
    for (int i = 1; i < 16; ++i) m = fmaxf(m, red[tid][i]);
    rowstat[tid] = m;
  }
  __syncthreads();
  {
    float m = rowstat[tl];
    float ps = 0.f;
    for (int k2 = kl * 16; k2 < kl * 16 + 16; ++k2) {
      float e2 = fast_exp2((sc[tl][k2] - m) * l2e);
      sc[tl][k2] = e2;
      ps += e2;
    }
    red[tl][kl] = ps;
  }
  __syncthreads();
  if (tid < 16) {
    float s = 0.f;
#pragma unroll
    for (int i = 0; i < 16; ++i) s += red[tid][i];
    rowstat[tid] = fast_rcp(s);
  }
  __syncthreads();
  {
    float inv = rowstat[tl];
    for (int k2 = kl * 16; k2 < kl * 16 + 16; ++k2)
      wbf[tl][k2] = f2bf(sc[tl][k2] * inv);
  }
  __syncthreads();

  // ---- Stage 3: context = w @ knowledge via v_wmma_f32_16x16x32_bf16 ----
  const int wave = tid >> 5;       // 0..7, each wave owns 2 N-tiles of 16
  const int lane = tid & 31;
  const int n16  = lane & 15;
  const int g    = lane >> 4;
  const unsigned int* kpB = kpack + (size_t)b * (8 * 16 * 2 * 16 * 8);

  for (int nt = wave * 2; nt < wave * 2 + 2; ++nt) {
    const int d0 = nt * 16;
    v8f acc = {};
#pragma unroll 2
    for (int kk = 0; kk < KK / 32; ++kk) {
      union { unsigned int w[8]; v16bf v; } Af, Bf;
      // B fragment: 8 consecutive words -> 2x global_load_b128
      const unsigned int* bp = kpB + ((size_t)(((kk * 16 + nt) * 2 + g) * 16 + n16)) * 8;
#pragma unroll
      for (int vv = 0; vv < 8; ++vv) Bf.w[vv] = bp[vv];
      // A fragment: 8 consecutive words (k = 16g + s) -> 2x ds_load_b128
#pragma unroll
      for (int vv = 0; vv < 8; ++vv) {
        int ka = kk * 32 + g * 16 + vv * 2;
        Af.w[vv] = *(const unsigned int*)&wbf[n16][ka];  // ka even -> aligned
      }
      acc = __builtin_amdgcn_wmma_f32_16x16x32_bf16(
          /*neg_a=*/false, Af.v, /*neg_b=*/false, Bf.v,
          /*c_mod=*/(short)0, acc, /*reuse_a=*/false, /*reuse_b=*/false);
    }
#pragma unroll
    for (int vv = 0; vv < 8; ++vv) {
      int m = vv + 8 * g;  // D layout: VGPR v -> M=v (lanes 0-15), M=v+8 (16-31)
      out[((size_t)(b * TT + t0 + m)) * DD + d0 + n16] = acc[vv];
    }
  }
}

extern "C" void kernel_launch(void* const* d_in, const int* in_sizes, int n_in,
                              void* d_out, int out_size, void* d_ws, size_t ws_size,
                              hipStream_t stream) {
  (void)in_sizes; (void)n_in; (void)out_size; (void)ws_size;
  const float* knowledge = (const float*)d_in[0];  // [B,K,D]
  const float* enc       = (const float*)d_in[1];  // [B,T,D_ENC]
  const float* W1        = (const float*)d_in[2];  // [D_ENC,U]
  const float* b1        = (const float*)d_in[3];  // [U]
  const float* W2        = (const float*)d_in[4];  // [D_K,U]
  const float* b2        = (const float*)d_in[5];  // [U]
  const float* Vv        = (const float*)d_in[6];  // [U,1]
  const float* bV        = (const float*)d_in[7];  // [1]
  float* out = (float*)d_out;

  float*        e_ws = (float*)d_ws;                       // [B,T,U] 1 MB
  float*        k_ws = e_ws + (size_t)BB * TT * UU;        // [B,K,U] 0.5 MB
  unsigned int* kp   = (unsigned int*)(k_ws + (size_t)BB * KK * UU);  // 0.5 MB

  // e = enc @ W1 + b1
  proj_kernel<<<(BB * TT * UU) / 256, 256, 0, stream>>>(enc, W1, b1, e_ws,
                                                        BB * TT, DD);
  // k = knowledge @ W2 + b2
  proj_kernel<<<(BB * KK * UU) / 256, 256, 0, stream>>>(knowledge, W2, b2, k_ws,
                                                        BB * KK, DD);
  // knowledge -> bf16 fragment-major pack
  repack_kernel<<<(BB * 8 * 16 * 2 * 16 * 8) / 256, 256, 0, stream>>>(knowledge, kp);
  // fused scores -> softmax -> WMMA context
  attn_fused_kernel<<<BB * (TT / 16), 256, 0, stream>>>(kp, e_ws, k_ws,
                                                        Vv, bV, out);
}
